// JacobianMLP_69157563400601
// MI455X (gfx1250) — matrix operand order repaired and checked by reading, not compiled, not run on hardware
//
#include <hip/hip_runtime.h>
#include <hip/hip_fp16.h>

typedef __attribute__((ext_vector_type(16))) _Float16 v16h;
typedef __attribute__((ext_vector_type(8)))  float    v8f;
typedef __attribute__((ext_vector_type(4)))  float    v4f;

#define B_TOT   32768
#define IN_DIM  16
#define OUT_DIM 16
#define HID     512

#define WAVES_PER_BLOCK 8
#define BATCH_PER_WAVE  16
#define BATCH_PER_BLOCK (WAVES_PER_BLOCK * BATCH_PER_WAVE)  // 128
#define NTILES          (HID / 16)                          // 32
#define PANEL_ELEMS     (NTILES * 32 * 16)                  // 16384 halfs = 32 KB

__global__ __launch_bounds__(256) void jacmlp_fused_kernel(
    const float* __restrict__ x,    // [B, I]
    const float* __restrict__ W1,   // [O, H, I]
    const float* __restrict__ b1,   // [O, H]
    const float* __restrict__ W2,   // [O, H]
    const float* __restrict__ b2,   // [O]
    float* __restrict__ out)        // [B, O]
{
    const int o    = blockIdx.y;
    const int tid  = threadIdx.x;
    const int lane = tid & 31;
    const int wave = tid >> 5;
    const int half = lane >> 4;
    const int l16  = lane & 15;

    // B-panel: per (tile, lane-slot) rows in exact WMMA-B order.
    //  slots 0-15 : K=0..15  -> W1[o][t*16+slot][0..15] (f16)
    //  slots 16-31: K=16..31 -> {bias(t*16+slot-16), 0 x15}
    __shared__ __align__(32) _Float16 bpanel[PANEL_ELEMS];  // 32 KB
    // W2 staged as (0.6*w2, 0.4*w2) pairs: w2*leaky(c) = 0.6w2*c + 0.4w2*|c|.
    __shared__ __align__(8) float w2p[HID * 2];             // 4 KB

    const float* W1o = W1 + (size_t)o * HID * IN_DIM;
    const float* b1o = b1 + o * HID;

    // --- Staging: branch-free, vectorized, 2 rows/thread of each kind. ---
    #pragma unroll
    for (int j = 0; j < 2; ++j) {
        const int row = tid * 2 + j;            // hidden unit 0..511
        const int t = row >> 4, ln = row & 15;
        const v4f* s4 = reinterpret_cast<const v4f*>(W1o + row * IN_DIM);
        const v4f f0 = s4[0], f1 = s4[1], f2 = s4[2], f3 = s4[3];
        v16h v;
        #pragma unroll
        for (int e = 0; e < 4; ++e) {
            v[e]      = (_Float16)f0[e];
            v[4 + e]  = (_Float16)f1[e];
            v[8 + e]  = (_Float16)f2[e];
            v[12 + e] = (_Float16)f3[e];
        }
        *reinterpret_cast<v16h*>(&bpanel[(t * 32 + ln) * 16]) = v;
    }
    #pragma unroll
    for (int j = 0; j < 2; ++j) {
        const int idx = tid * 2 + j;            // bias slot 0..511
        const int t = idx >> 4, ln2 = idx & 15;
        v16h v = {};
        v[0] = (_Float16)b1o[t * 16 + ln2];     // K=16 row: bias
        *reinterpret_cast<v16h*>(&bpanel[(t * 32 + 16 + ln2) * 16]) = v;
    }
    #pragma unroll
    for (int j = 0; j < 2; ++j) {
        const int h = tid * 2 + j;
        const float w = W2[o * HID + h];
        w2p[h * 2]     = 0.6f * w;
        w2p[h * 2 + 1] = 0.4f * w;
    }
    __syncthreads();

    const int bbase = blockIdx.x * BATCH_PER_BLOCK + wave * BATCH_PER_WAVE;

    // A operand (contiguous K=16h+s): half 0 = X row K0..15, half 1 = {1.0 (bias, K=16), 0 x15}.
    v16h a = {};
    if (half == 0) {
        const v4f* xp = reinterpret_cast<const v4f*>(
            x + (size_t)(bbase + l16) * IN_DIM);
        const v4f x0 = xp[0], x1 = xp[1], x2 = xp[2], x3 = xp[3];
        #pragma unroll
        for (int e = 0; e < 4; ++e) {
            a[e]      = (_Float16)x0[e];
            a[4 + e]  = (_Float16)x1[e];
            a[8 + e]  = (_Float16)x2[e];
            a[12 + e] = (_Float16)x3[e];
        }
    } else {
        a[0] = (_Float16)1.0f;
    }

    float acc[8];
    #pragma unroll
    for (int r = 0; r < 8; ++r) acc[r] = 0.0f;

    const _Float16* mypanel = bpanel + lane * 16;
    const float*    myw2    = w2p + l16 * 2;
    const v8f czero = {};

    auto panel = [&](int t) -> v16h {
        return *reinterpret_cast<const v16h*>(mypanel + t * (32 * 16));
    };
    auto w2at = [&](int t) -> float2 {
        return *reinterpret_cast<const float2*>(myw2 + t * 32);
    };
    auto epi = [&](const v8f& c, const float2& w) {
        #pragma unroll
        for (int r = 0; r < 8; ++r)
            acc[r] = fmaf(fabsf(c[r]), w.y, fmaf(c[r], w.x, acc[r]));
    };

    // Groups of 4 tiles with 4 independent C register sets: each WMMA's
    // consumers are >= 3 matrix ops / >= 12 VALU away, so the WMMA->VALU
    // co-exec hazard slots fill with real work instead of v_nops.
    #pragma unroll
    for (int g = 0; g < NTILES; g += 4) {
        const v16h bm0 = panel(g + 0);
        const v16h bm1 = panel(g + 1);
        const v16h bm2 = panel(g + 2);
        const v16h bm3 = panel(g + 3);

        const v8f c0 = __builtin_amdgcn_wmma_f32_16x16x32_f16(
            false, a, false, bm0, (short)0, czero, false, false);
        const v8f c1 = __builtin_amdgcn_wmma_f32_16x16x32_f16(
            false, a, false, bm1, (short)0, czero, false, false);
        const v8f c2 = __builtin_amdgcn_wmma_f32_16x16x32_f16(
            false, a, false, bm2, (short)0, czero, false, false);
        const v8f c3 = __builtin_amdgcn_wmma_f32_16x16x32_f16(
            false, a, false, bm3, (short)0, czero, false, false);

        const float2 w0 = w2at(g + 0);
        const float2 w1 = w2at(g + 1);
        const float2 w2v = w2at(g + 2);
        const float2 w3 = w2at(g + 3);

        epi(c0, w0);
        epi(c1, w1);
        epi(c2, w2v);
        epi(c3, w3);
    }

    // Reduce over the 16 N-lanes within each half (wave32 xor-shuffles).
    #pragma unroll
    for (int r = 0; r < 8; ++r) {
        #pragma unroll
        for (int off = 1; off < 16; off <<= 1)
            acc[r] += __shfl_xor(acc[r], off, 32);
    }

    if (l16 < 8) {
        float val = acc[0];
        #pragma unroll
        for (int j = 1; j < 8; ++j)
            if (l16 == j) val = acc[j];
        const int m = half * 8 + l16;
        out[(size_t)(bbase + m) * OUT_DIM + o] = val + b2[o];
    }
}

extern "C" void kernel_launch(void* const* d_in, const int* in_sizes, int n_in,
                              void* d_out, int out_size, void* d_ws, size_t ws_size,
                              hipStream_t stream) {
    (void)in_sizes; (void)n_in; (void)out_size; (void)d_ws; (void)ws_size;
    const float* x  = (const float*)d_in[0];
    const float* W1 = (const float*)d_in[1];
    const float* b1 = (const float*)d_in[2];
    const float* W2 = (const float*)d_in[3];
    const float* b2 = (const float*)d_in[4];
    float* out = (float*)d_out;

    dim3 grid(B_TOT / BATCH_PER_BLOCK, OUT_DIM);  // (256, 16)
    dim3 block(256);
    jacmlp_fused_kernel<<<grid, block, 0, stream>>>(x, W1, b1, W2, b2, out);
}
